// SpanClassifier_2422361555735
// MI455X (gfx1250) — matrix operand order repaired and checked by reading, not compiled, not run on hardware
//
#include <hip/hip_runtime.h>
#include <math.h>

// Problem constants (match reference)
#define BB 4
#define LL 1024
#define DD 768
#define RR 16
#define SS 8
#define NREL 33          // 2R+1
#define RELP 48          // padded rel columns (3 N-tiles)

typedef __attribute__((ext_vector_type(16))) _Float16 v16h;
typedef __attribute__((ext_vector_type(8)))  _Float16 v8h;
typedef __attribute__((ext_vector_type(8)))  float    v8f;

// ---------------------------------------------------------------------------
// Core: one 16x(16*NT) f32 strip of  C = A(MxK) * B(KxN); f16 ops, K%32==0.
// Compile-time LDA/LDB so the K-loop strength-reduces to constant pointer
// increments (no per-iteration 64-bit multiplies).
//   Arow: &A[(m0+(lane&15))*LDA + khalf]   (khalf = 8 for lanes 16..31)
//   Bl:   &B[lane*LDB + n0]                (lane = K row within 32-block)
// A-frag (16-bit 16x32): lanes 0-15 M=lane, K {0..7,16..23}; lanes 16-31
// same M, K {8..15,24..31}.  B-frag (32x16): N = lane&15, 16 halves = K rows in A's order.
// ---------------------------------------------------------------------------
template <int LDA, int LDB, int NT>
__device__ __forceinline__ void wmma_strip(const _Float16* __restrict__ Arow,
                                           const _Float16* __restrict__ Bl,
                                           int K, v8f (&acc)[NT]) {
#pragma unroll 2
  for (int k0 = 0; k0 < K; k0 += 32) {
    v8h lo = *(const v8h*)(Arow);
    v8h hi = *(const v8h*)(Arow + 16);
    v16h a;
#pragma unroll
    for (int j = 0; j < 8; ++j) { a[j] = lo[j]; a[j + 8] = hi[j]; }
#pragma unroll
    for (int t = 0; t < NT; ++t) {
      v16h b;
      const int ln = threadIdx.x & 31, kh = (ln & 16) >> 1;
      const _Float16* Bc = Bl - (size_t)ln * LDB + (ln & 15) + t * 16;
#pragma unroll
      for (int j = 0; j < 8; ++j) {
        b[j] = Bc[(size_t)(kh + j) * LDB];
        b[j + 8] = Bc[(size_t)(16 + kh + j) * LDB];
      }
      acc[t] = __builtin_amdgcn_wmma_f32_16x16x32_f16(false, a, false, b,
                                                      (short)0, acc[t],
                                                      false, false);
    }
    Arow += 32;
    Bl += (size_t)32 * LDB;
    // speculative DEV-scope prefetch ~3 K-steps ahead (safe past end: dropped)
    __builtin_prefetch((const void*)(Arow + 64), 0, 1);
    __builtin_prefetch((const void*)(Bl + (size_t)64 * LDB), 0, 1);
  }
}

// ---------------------------------------------------------------------------
// Generic GEMM, f16 output, optional bias + scale:  Out = (A*B + bias)*scale
// grid: x = Ntiles/(4*NT) [4 waves/block], y = Mtiles, z = batch
// ---------------------------------------------------------------------------
template <int LDA, int LDB, int NT, int LDO, int NWAVE>
__global__ void gemm_f16_k(const _Float16* __restrict__ A, size_t aStr,
                           const _Float16* __restrict__ Bm, size_t bStr,
                           _Float16* __restrict__ Out, size_t oStr,
                           const float* __restrict__ bias, float scale,
                           int K) {
  const int lane = threadIdx.x & 31, wv = threadIdx.x >> 5;
  const int nj = blockIdx.x * NWAVE + wv;
  const int mt = blockIdx.y, b = blockIdx.z;
  const int m0 = mt * 16, n0 = nj * 16 * NT;
  const int khalf = (lane & 16) >> 1;
  const _Float16* Arow =
      A + (size_t)b * aStr + (size_t)(m0 + (lane & 15)) * LDA + khalf;
  const _Float16* Bl = Bm + (size_t)b * bStr + (size_t)lane * LDB + n0;
  v8f acc[NT] = {};
  wmma_strip<LDA, LDB, NT>(Arow, Bl, K, acc);
  const int mb = m0 + ((lane & 16) >> 1);
  _Float16* o = Out + (size_t)b * oStr;
#pragma unroll
  for (int t = 0; t < NT; ++t) {
    const int n = n0 + t * 16 + (lane & 15);
    const float bv = bias ? bias[n] : 0.0f;
#pragma unroll
    for (int r = 0; r < 8; ++r)
      o[(size_t)(mb + r) * LDO + n] = (_Float16)((acc[t][r] + bv) * scale);
  }
}

// Generic GEMM, f32 output, no bias (t = q @ relT). Single wave per block.
template <int LDA, int LDB, int NT, int LDO>
__global__ void gemm_f32_k(const _Float16* __restrict__ A, size_t aStr,
                           const _Float16* __restrict__ Bm,
                           float* __restrict__ Out, size_t oStr, int K) {
  const int lane = threadIdx.x & 31;
  const int mt = blockIdx.y, b = blockIdx.z;
  const int m0 = mt * 16;
  const int khalf = (lane & 16) >> 1;
  const _Float16* Arow =
      A + (size_t)b * aStr + (size_t)(m0 + (lane & 15)) * LDA + khalf;
  const _Float16* Bl = Bm + (size_t)lane * LDB;
  v8f acc[NT] = {};
  wmma_strip<LDA, LDB, NT>(Arow, Bl, K, acc);
  const int mb = m0 + ((lane & 16) >> 1);
  float* o = Out + (size_t)b * oStr;
#pragma unroll
  for (int t = 0; t < NT; ++t) {
    const int n = t * 16 + (lane & 15);
#pragma unroll
    for (int r = 0; r < 8; ++r) o[(size_t)(mb + r) * LDO + n] = acc[t][r];
  }
}

// k = hid@Wk + bk, stored TRANSPOSED as kT[b][d][token] (f16) for coalesced
// B-fragment loads in the scores GEMM. grid: x=3, y=64, z=B, block 128.
__global__ void gemm_kT(const _Float16* __restrict__ hidH,
                        const _Float16* __restrict__ W,
                        const float* __restrict__ bias,
                        _Float16* __restrict__ kT) {
  const int lane = threadIdx.x & 31, wv = threadIdx.x >> 5;
  const int nj = blockIdx.x * 4 + wv;
  const int mt = blockIdx.y, b = blockIdx.z;
  const int m0 = mt * 16, n0 = nj * 64;
  const int khalf = (lane & 16) >> 1;
  const _Float16* Arow =
      hidH + (size_t)b * LL * DD + (size_t)(m0 + (lane & 15)) * DD + khalf;
  const _Float16* Bl = W + (size_t)lane * DD + n0;
  v8f acc[4] = {};
  wmma_strip<DD, DD, 4>(Arow, Bl, DD, acc);
  const int mb = m0 + ((lane & 16) >> 1);  // token index (8 consecutive)
#pragma unroll
  for (int t = 0; t < 4; ++t) {
    const int n = n0 + t * 16 + (lane & 15);  // d index
    const float bv = bias[n];
    v8h v;
#pragma unroll
    for (int r = 0; r < 8; ++r) v[r] = (_Float16)(acc[t][r] + bv);
    *(v8h*)(kT + (size_t)b * DD * LL + (size_t)n * LL + mb) = v;
  }
}

// scores[b,m,n] = q[b,m,:].kT[b,:,n] + t[b,m,ridx(m,n)], masked to -1e18.
// grid: x=4 (64 N-tiles / (4 waves * 4 tiles)), y=64, z=B, block 128.
__global__ void gemm_scores(const _Float16* __restrict__ q,
                            const _Float16* __restrict__ kT,
                            const float* __restrict__ t,
                            const int* __restrict__ sp,
                            const float* __restrict__ att,
                            float* __restrict__ scores, int iter) {
  const int lane = threadIdx.x & 31, wv = threadIdx.x >> 5;
  const int nj = blockIdx.x * 4 + wv;
  const int mt = blockIdx.y, b = blockIdx.z;
  const int m0 = mt * 16, n0 = nj * 64;
  const int khalf = (lane & 16) >> 1;
  const _Float16* Arow =
      q + (size_t)b * LL * DD + (size_t)(m0 + (lane & 15)) * DD + khalf;
  const _Float16* Bl = kT + (size_t)b * DD * LL + (size_t)lane * LL + n0;
  v8f acc[4] = {};
  wmma_strip<DD, LL, 4>(Arow, Bl, DD, acc);
  const int mb = m0 + ((lane & 16) >> 1);
#pragma unroll
  for (int t4 = 0; t4 < 4; ++t4) {
    const int n = n0 + t4 * 16 + (lane & 15);
    const float am = att[b * LL + n];
#pragma unroll
    for (int r = 0; r < 8; ++r) {
      const int m = mb + r;
      int d = n - m;
      d = (d < -RR ? -RR : (d > RR ? RR : d)) + RR;
      const float s = acc[t4][r] + t[(size_t)(b * LL + m) * RELP + d];
      const bool keep = (iter < sp[b * LL + m]) && (am > 0.0f);
      scores[(size_t)(b * LL + m) * LL + n] = keep ? s : -1e18f;
    }
  }
}

// Row softmax over L=1024; writes f32 probs to d_out slice [row, iter, :]
// and an f16 copy for the following probs@hid GEMM.
__global__ void softmax_rows(const float* __restrict__ scores,
                             float* __restrict__ outp,
                             _Float16* __restrict__ aP, int iter) {
  const int row = blockIdx.x;  // b*L + m
  const float* s = scores + (size_t)row * LL;
  __shared__ float red[8];
  const int tid = threadIdx.x, lane = tid & 31, wv = tid >> 5;

  float mx = -3.4e38f;
  for (int j = tid; j < LL; j += 256) mx = fmaxf(mx, s[j]);
#pragma unroll
  for (int o = 16; o > 0; o >>= 1) mx = fmaxf(mx, __shfl_xor(mx, o, 32));
  if (lane == 0) red[wv] = mx;
  __syncthreads();
  mx = red[0];
#pragma unroll
  for (int w = 1; w < 8; ++w) mx = fmaxf(mx, red[w]);
  __syncthreads();

  float sum = 0.0f;
  for (int j = tid; j < LL; j += 256) sum += __expf(s[j] - mx);
#pragma unroll
  for (int o = 16; o > 0; o >>= 1) sum += __shfl_xor(sum, o, 32);
  if (lane == 0) red[wv] = sum;
  __syncthreads();
  sum = 0.0f;
#pragma unroll
  for (int w = 0; w < 8; ++w) sum += red[w];
  const float inv = 1.0f / sum;

  float* orow = outp + ((size_t)row * SS + iter) * LL;
  _Float16* arow = aP + (size_t)row * LL;
  for (int j = tid; j < LL; j += 256) {
    const float p = __expf(s[j] - mx) * inv;
    orow[j] = p;
    arow[j] = (_Float16)p;
  }
}

__global__ void f32_to_f16(const float* __restrict__ in,
                           _Float16* __restrict__ out, int n) {
  const int i = blockIdx.x * 256 + threadIdx.x;
  if (i < n) out[i] = (_Float16)in[i];
}

// rel [33, D] f32 -> relT [D, 48] f16 (zero-padded columns 33..47)
__global__ void build_relT(const float* __restrict__ rel,
                           _Float16* __restrict__ relT) {
  const int i = blockIdx.x * 256 + threadIdx.x;
  if (i >= DD * RELP) return;
  const int r = i % RELP, d = i / RELP;
  relT[i] = (r < NREL) ? (_Float16)rel[r * DD + d] : (_Float16)0.0f;
}

__global__ void write_masks(const int* __restrict__ sp,
                            float* __restrict__ outm) {
  const int idx = blockIdx.x * 256 + threadIdx.x;  // over B*L*S
  if (idx >= BB * LL * SS) return;
  const int i = idx % SS, row = idx / SS;
  outm[idx] = (i < sp[row]) ? 1.0f : 0.0f;
}

// ---------------------------------------------------------------------------
extern "C" void kernel_launch(void* const* d_in, const int* in_sizes, int n_in,
                              void* d_out, int out_size, void* d_ws,
                              size_t ws_size, hipStream_t stream) {
  const float* hid   = (const float*)d_in[0];
  const int*   sp    = (const int*)d_in[1];
  const float* att   = (const float*)d_in[3];
  const float* Wq_st = (const float*)d_in[4];
  const float* bq_st = (const float*)d_in[5];
  const float* Wk_st = (const float*)d_in[6];
  const float* bk_st = (const float*)d_in[7];
  const float* relst = (const float*)d_in[8];
  const float* Wq_ed = (const float*)d_in[9];
  const float* bq_ed = (const float*)d_in[10];
  const float* Wk_ed = (const float*)d_in[11];
  const float* bk_ed = (const float*)d_in[12];
  const float* reled = (const float*)d_in[13];
  const float* Wsp   = (const float*)d_in[14];
  const float* bsp   = (const float*)d_in[15];

  // workspace carve-out (256B aligned blocks)
  char* base = (char*)d_ws;
  size_t off = 0;
  auto carve = [&](size_t bytes) {
    char* p = base + off;
    off = (off + bytes + 255) & ~(size_t)255;
    return p;
  };
  const size_t HIDB = (size_t)BB * LL * DD * 2;  // f16 [B,L,D]
  const size_t WB   = (size_t)DD * DD * 2;       // f16 [D,D]
  _Float16* hidH   = (_Float16*)carve(HIDB);
  _Float16* wqstH  = (_Float16*)carve(WB);
  _Float16* wkstH  = (_Float16*)carve(WB);
  _Float16* wqedH  = (_Float16*)carve(WB);
  _Float16* wkedH  = (_Float16*)carve(WB);
  _Float16* wspH   = (_Float16*)carve(WB);
  _Float16* relstT = (_Float16*)carve((size_t)DD * RELP * 2);
  _Float16* reledT = (_Float16*)carve((size_t)DD * RELP * 2);
  _Float16* kTst   = (_Float16*)carve(HIDB);     // [B,D,L]
  _Float16* kTed   = (_Float16*)carve(HIDB);
  _Float16* qH     = (_Float16*)carve(HIDB);
  float*    tB     = (float*)carve((size_t)BB * LL * RELP * 4);
  float*    sc     = (float*)carve((size_t)BB * LL * LL * 4);
  _Float16* aH     = (_Float16*)carve((size_t)BB * LL * LL * 2);
  _Float16* updH   = (_Float16*)carve(HIDB);
  _Float16* h1     = (_Float16*)carve(HIDB);
  _Float16* h2     = (_Float16*)carve(HIDB);
  (void)ws_size; (void)in_sizes; (void)n_in; (void)out_size;

  float* outF = (float*)d_out;
  float* sts  = outF;
  float* eds  = outF + (size_t)BB * LL * SS * LL;
  float* msk  = outF + 2 * (size_t)BB * LL * SS * LL;

  // ---- setup: f16 conversions + rel transposes + kT GEMMs ----
  const int NH = BB * LL * DD, NW = DD * DD;
  f32_to_f16<<<(NH + 255) / 256, 256, 0, stream>>>(hid, hidH, NH);
  f32_to_f16<<<(NW + 255) / 256, 256, 0, stream>>>(Wq_st, wqstH, NW);
  f32_to_f16<<<(NW + 255) / 256, 256, 0, stream>>>(Wk_st, wkstH, NW);
  f32_to_f16<<<(NW + 255) / 256, 256, 0, stream>>>(Wq_ed, wqedH, NW);
  f32_to_f16<<<(NW + 255) / 256, 256, 0, stream>>>(Wk_ed, wkedH, NW);
  f32_to_f16<<<(NW + 255) / 256, 256, 0, stream>>>(Wsp, wspH, NW);
  build_relT<<<(DD * RELP + 255) / 256, 256, 0, stream>>>(relst, relstT);
  build_relT<<<(DD * RELP + 255) / 256, 256, 0, stream>>>(reled, reledT);
  gemm_kT<<<dim3(3, 64, BB), 128, 0, stream>>>(hidH, wkstH, bk_st, kTst);
  gemm_kT<<<dim3(3, 64, BB), 128, 0, stream>>>(hidH, wkedH, bk_ed, kTed);

  const float invs = 1.0f / sqrtf((float)DD);
  const size_t SLD = (size_t)LL * DD;   // per-batch stride [L,D]
  const size_t SLL = (size_t)LL * LL;   // per-batch stride [L,L]

  for (int i = 0; i < SS; ++i) {
    // ---------------- start-pointer branch ----------------
    const _Float16* qsrc = (i == 0) ? hidH : h1;
    gemm_f16_k<DD, DD, 4, DD, 4><<<dim3(3, 64, BB), 128, 0, stream>>>(
        qsrc, SLD, wqstH, 0, qH, SLD, bq_st, invs, DD);
    gemm_f32_k<DD, RELP, 3, RELP><<<dim3(1, 64, BB), 32, 0, stream>>>(
        qH, SLD, relstT, tB, (size_t)LL * RELP, DD);
    gemm_scores<<<dim3(4, 64, BB), 128, 0, stream>>>(qH, kTst, tB, sp, att,
                                                     sc, i);
    softmax_rows<<<dim3(BB * LL), 256, 0, stream>>>(sc, sts, aH, i);
    gemm_f16_k<LL, DD, 4, DD, 4><<<dim3(3, 64, BB), 128, 0, stream>>>(
        aH, SLL, hidH, SLD, updH, SLD, nullptr, 1.0f, LL);
    gemm_f16_k<DD, DD, 4, DD, 4><<<dim3(3, 64, BB), 128, 0, stream>>>(
        updH, SLD, wspH, 0, h1, SLD, bsp, 1.0f, DD);

    // ---------------- end-pointer branch ----------------
    const _Float16* qsrc2 = (i == 0) ? hidH : h2;
    gemm_f16_k<DD, DD, 4, DD, 4><<<dim3(3, 64, BB), 128, 0, stream>>>(
        qsrc2, SLD, wqedH, 0, qH, SLD, bq_ed, invs, DD);
    gemm_f32_k<DD, RELP, 3, RELP><<<dim3(1, 64, BB), 32, 0, stream>>>(
        qH, SLD, reledT, tB, (size_t)LL * RELP, DD);
    gemm_scores<<<dim3(4, 64, BB), 128, 0, stream>>>(qH, kTed, tB, sp, att,
                                                     sc, i);
    softmax_rows<<<dim3(BB * LL), 256, 0, stream>>>(sc, eds, aH, i);
    gemm_f16_k<LL, DD, 4, DD, 4><<<dim3(3, 64, BB), 128, 0, stream>>>(
        aH, SLL, hidH, SLD, updH, SLD, nullptr, 1.0f, LL);
    gemm_f16_k<DD, DD, 4, DD, 4><<<dim3(3, 64, BB), 128, 0, stream>>>(
        updH, SLD, wspH, 0, h2, SLD, bsp, 1.0f, DD);
  }

  write_masks<<<(BB * LL * SS + 255) / 256, 256, 0, stream>>>(sp, msk);
}
